// Decoder_7464653160894
// MI455X (gfx1250) — compile-verified, matrix operand order checked by reading, not run on hardware
//
#include <hip/hip_runtime.h>
#include <cstdint>

constexpr int T  = 150;
constexpr int TP = 160;     // T padded to 10 WMMA row-tiles
constexpr int D  = 64;
constexpr int H  = 8;
constexpr int HD = 8;
constexpr int A  = 37;
constexpr int FF = 256;
constexpr int NTHREADS = 256;   // 8 wave32

typedef __attribute__((ext_vector_type(16))) __bf16 v16bf;
typedef __attribute__((ext_vector_type(8)))  __bf16 v8bf;
typedef __attribute__((ext_vector_type(8)))  float  v8f;
typedef __attribute__((ext_vector_type(4)))  unsigned int u32x4;
typedef __attribute__((ext_vector_type(8)))  int i32x8;
typedef __attribute__((ext_vector_type(4)))  int i32x4;

struct DecP {
  const float* enc; const float* x1; const float* x2; const float* x3;
  const int* mask; const int* y;
  const float* conv1_w; const float* conv1_b;
  const float* conv2_w; const float* conv2_b;
  const float* fc_x1_w; const float* fc_x1_b;
  const float* fc_m1_w; const float* fc_m1_b;
  const float* fc_m2_w; const float* fc_m2_b;
  const float* sa_wq; const float* sa_bq; const float* sa_bv;
  const float* sa_pw; const float* sa_pb;
  const float* ca_wq; const float* ca_bq; const float* ca_bv;
  const float* ca_pw; const float* ca_pb;
  const float* ff_w1; const float* ff_b1; const float* ff_w2; const float* ff_b2;
  const float* head_w; const float* head_b;
  const float* ln1_g; const float* ln1_b;
  const float* ln2_g; const float* ln2_b;
  const float* ln3_g; const float* ln3_b;
  const float* lnf_g; const float* lnf_b;
  const unsigned short* wbf;   // 4x [64][64] bf16 bits, [n][k] layout
  float* out_logits; float* losses;
};

__device__ __forceinline__ unsigned short f2bf(float f){
  unsigned int u = __float_as_uint(f);
  return (unsigned short)((u + 0x7FFFu + ((u >> 16) & 1u)) >> 16);  // RNE
}

// LayerNorm over 64 elems (per-row scalar stats: negligible cost).
__device__ __forceinline__ void ln64(int tid, const float* g, const float* bb,
                                     const float* x, float* out, float* stats){
  if (tid == 0){
    float m = 0.f;
    for (int i = 0; i < D; ++i) m += x[i];
    m *= (1.f/D);
    float v = 0.f;
    for (int i = 0; i < D; ++i){ float d = x[i]-m; v += d*d; }
    stats[0] = m; stats[1] = rsqrtf(v*(1.f/D) + 1e-5f);
  }
  __syncthreads();
  if (tid < D) out[tid] = (x[tid]-stats[0])*stats[1]*g[tid] + bb[tid];
  __syncthreads();
}

// Single-query MHA over enc (in LDS as bf16). K/V projections on WMMA.
// bk dropped (per-head softmax shift invariance); bv folded in post-sum.
__device__ __forceinline__ void mha1q(
    int tid,
    const float* wq, const float* bq, const float* bv,
    const float* pw, const float* pb,
    const unsigned short* wtk, const unsigned short* wtv,
    const unsigned short* encB, float* Kmat, float* Vmat,
    float* scores, float* qv, float* attout, float* hsum,
    float* xvec, const float* xn)
{
  if (tid < D){
    float acc = bq[tid];
    for (int i = 0; i < D; ++i) acc += xn[i]*wq[i*D+tid];
    qv[tid] = acc * 0.35355339059f;   // 1/sqrt(HD)
  }
  __syncthreads();

  // ---- WMMA: K = enc@wk, V = enc@wv; 10x4 tiles each, 80 tiles over 8 waves
  {
    const int wave = tid >> 5, lane = tid & 31, lh = lane & 15, hi = lane >> 4;
    for (int tile = wave; tile < 80; tile += 8){
      const int isv = tile >= 40;
      const int t2  = tile - (isv ? 40 : 0);
      const int mt  = t2 >> 2, nt = t2 & 3;
      const unsigned short* wt = isv ? wtv : wtk;
      float* op = isv ? Vmat : Kmat;
      v8f acc = {};
#pragma unroll
      for (int k0 = 0; k0 < D; k0 += 32){
        // A fragment: 16-bit A 16x32 lane layout (lane<16: K lo-half; lane>=16: +8)
        const int arow = (mt*16 + lh)*D + k0 + hi*8;
        v8bf a0 = *reinterpret_cast<const v8bf*>(&encB[arow]);
        v8bf a1 = *reinterpret_cast<const v8bf*>(&encB[arow + 16]);
        v16bf av = __builtin_shufflevector(a0, a1, 0,1,2,3,4,5,6,7,8,9,10,11,12,13,14,15);
        // B fragment: weights prepacked [n][k] bf16 -> contiguous 16B chunks
        const int brow = (nt*16 + lh)*D + k0 + hi*8;
        v8bf b0 = *reinterpret_cast<const v8bf*>(&wt[brow]);
        v8bf b1 = *reinterpret_cast<const v8bf*>(&wt[brow + 16]);
        v16bf bm = __builtin_shufflevector(b0, b1, 0,1,2,3,4,5,6,7,8,9,10,11,12,13,14,15);
        acc = __builtin_amdgcn_wmma_f32_16x16x32_bf16(false, av, false, bm,
                                                      (short)0, acc, false, false);
      }
#pragma unroll
      for (int r = 0; r < 8; ++r)
        op[(mt*16 + hi*8 + r)*D + nt*16 + lh] = acc[r];   // C/D layout: M=r(+8*hi), N=lane&15
    }
  }
  __syncthreads();

  // scores s[h][t] = q_h . k_t   (q pre-scaled)
  for (int i = tid; i < H*T; i += NTHREADS){
    int h = i / T, t = i % T;
    float s = 0.f;
    for (int dd = 0; dd < HD; ++dd) s += qv[h*HD+dd]*Kmat[t*D + h*HD + dd];
    scores[h*TP + t] = s;
  }
  __syncthreads();

  // softmax: one wave per head, shfl_xor reductions (wave32)
  {
    const int wave = tid >> 5, lane = tid & 31;
    float m = -1e30f;
    for (int t = lane; t < T; t += 32) m = fmaxf(m, scores[wave*TP + t]);
    for (int off = 16; off; off >>= 1) m = fmaxf(m, __shfl_xor(m, off, 32));
    float s = 0.f;
    for (int t = lane; t < T; t += 32){
      float e = __expf(scores[wave*TP + t] - m);
      scores[wave*TP + t] = e; s += e;
    }
    for (int off = 16; off; off >>= 1) s += __shfl_xor(s, off, 32);
    if (lane == 0) hsum[wave] = s;
  }
  __syncthreads();

  if (tid < D){
    int h = tid >> 3;
    float acc = 0.f;
    for (int t = 0; t < T; ++t) acc += scores[h*TP + t]*Vmat[t*D + tid];
    attout[tid] = acc / hsum[h] + bv[tid];
  }
  __syncthreads();
  if (tid < D){
    float acc = pb[tid];
    for (int i = 0; i < D; ++i) acc += attout[i]*pw[i*D+tid];
    xvec[tid] += acc;
  }
  __syncthreads();
}

__global__ __launch_bounds__(NTHREADS)
void decoder_fused(DecP P, int Bn){
  __shared__ unsigned short encB[TP*D];   // 20 KB bf16
  __shared__ float Kmat[TP*D];            // 40 KB (also fp32 TDM staging)
  __shared__ float Vmat[TP*D];            // 40 KB
  __shared__ float scores[H*TP];          // 5 KB (reused as logits)
  __shared__ float conv1h[8*37];
  __shared__ float conv2h[16*37];
  __shared__ float xvec[D], xn[D], qv[D], attout[D], h1s[D];
  __shared__ float ffh[FF];
  __shared__ float hsum[H];
  __shared__ float stats[2];

  const int tid = threadIdx.x;
  const int b = blockIdx.x;
  if (b >= Bn) return;

  // ---- TDM: async DMA enc_out[b] (150x64 fp32 = 38400 B) -> LDS (Kmat region).
  // Issued once (wave 0) at kernel start; overlaps with the state net below.
  const float* enc = P.enc + (size_t)b*T*D;
  float* encF = Kmat;                      // fp32 staging aliases Kmat
  if (tid < 32){
    const uint64_t ga  = (uint64_t)(uintptr_t)enc;
    const unsigned lds = (unsigned)(uintptr_t)encF;  // flat->LDS: low 32 bits
    // D# group0: count=1, lds_addr, global_addr[56:0], type=2
    u32x4 g0 = { 1u, lds, (unsigned)(ga & 0xffffffffu),
                 (unsigned)((ga >> 32) & 0x1ffffffu) | 0x80000000u };
    // D# group1: data_size=4B; tensor/tile = one row of 9600 elements
    // [17:16]=2 | dim0=9600 | dim1=1 | tile0=9600 | tile1=1 | stride0=9600
    i32x8 g1 = { 0x00020000, 0x25800000, 0x00010000, 0x25800000,
                 1, 9600, 0, 0 };
    i32x4 gz = { 0, 0, 0, 0 };
#if __clang_major__ >= 23
    i32x8 gz8 = { 0, 0, 0, 0, 0, 0, 0, 0 };
    __builtin_amdgcn_tensor_load_to_lds(g0, g1, gz, gz, gz8, 0);
#else
    __builtin_amdgcn_tensor_load_to_lds(g0, g1, gz, gz, 0);
#endif
  }

  // ---- state net (independent of enc; runs under the TDM copy) ----
  const float* x1 = P.x1 + (size_t)b*3*37;
  for (int i = tid; i < 8*37; i += NTHREADS){
    int o = i/37, pos = i%37;
    float acc = P.conv1_b[o];
    for (int c = 0; c < 3; ++c)
      for (int kk = 0; kk < 3; ++kk){
        int xi = pos + kk - 1;
        if (xi >= 0 && xi < 37) acc += P.conv1_w[o*9 + c*3 + kk]*x1[c*37 + xi];
      }
    conv1h[i] = fmaxf(acc, 0.f);
  }
  __syncthreads();
  for (int i = tid; i < 16*37; i += NTHREADS){
    int o = i/37, pos = i%37;
    float acc = P.conv2_b[o];
    for (int c = 0; c < 8; ++c)
      for (int kk = 0; kk < 3; ++kk){
        int xi = pos + kk - 1;
        if (xi >= 0 && xi < 37) acc += P.conv2_w[o*24 + c*3 + kk]*conv1h[c*37 + xi];
      }
    conv2h[i] = fmaxf(acc, 0.f);
  }
  __syncthreads();
  if (tid < D){
    float acc = P.fc_x1_b[tid];
    for (int i = 0; i < 592; ++i) acc += conv2h[i]*P.fc_x1_w[i*D + tid];
    for (int i = 0; i < 111; ++i) acc += x1[i]*P.fc_x1_w[(592+i)*D + tid];
    h1s[tid] = fmaxf(acc, 0.f);
  }
  __syncthreads();
  if (tid < D){
    float acc = P.fc_m1_b[tid];
    for (int i = 0; i < D; ++i) acc += h1s[i]*P.fc_m1_w[i*D + tid];
    const float* x2 = P.x2 + (size_t)b*28;
    for (int i = 0; i < 28; ++i) acc += x2[i]*P.fc_m1_w[(64+i)*D + tid];
    const float* x3 = P.x3 + (size_t)b*4;
    for (int i = 0; i < 4; ++i) acc += x3[i]*P.fc_m1_w[(92+i)*D + tid];
    attout[tid] = fmaxf(acc, 0.f);       // temp h2
  }
  __syncthreads();
  if (tid < D){
    float acc = P.fc_m2_b[tid];
    for (int i = 0; i < D; ++i) acc += attout[i]*P.fc_m2_w[i*D + tid];
    xvec[tid] = fmaxf(acc, 0.f);
  }

  // ---- drain TDM, then convert fp32 staging -> bf16 tiles ----
  if (tid < 32) __builtin_amdgcn_s_wait_tensorcnt(0);
  __syncthreads();
  for (int i = tid; i < TP*D; i += NTHREADS){
    int row = i >> 6;
    encB[i] = f2bf(row < T ? encF[i] : 0.f);
  }
  __syncthreads();

  // ---- decoder block ----
  ln64(tid, P.ln1_g, P.ln1_b, xvec, xn, stats);
  mha1q(tid, P.sa_wq, P.sa_bq, P.sa_bv, P.sa_pw, P.sa_pb,
        P.wbf + 0, P.wbf + 4096, encB, Kmat, Vmat, scores, qv, attout, hsum, xvec, xn);
  ln64(tid, P.ln2_g, P.ln2_b, xvec, xn, stats);
  mha1q(tid, P.ca_wq, P.ca_bq, P.ca_bv, P.ca_pw, P.ca_pb,
        P.wbf + 8192, P.wbf + 12288, encB, Kmat, Vmat, scores, qv, attout, hsum, xvec, xn);
  ln64(tid, P.ln3_g, P.ln3_b, xvec, xn, stats);

  for (int j = tid; j < FF; j += NTHREADS){
    float acc = P.ff_b1[j];
    for (int i = 0; i < D; ++i) acc += xn[i]*P.ff_w1[i*FF + j];
    ffh[j] = 0.5f*acc*(1.f + erff(acc*0.70710678118f));   // exact GELU
  }
  __syncthreads();
  if (tid < D){
    float acc = P.ff_b2[tid];
    for (int j = 0; j < FF; ++j) acc += ffh[j]*P.ff_w2[j*D + tid];
    xvec[tid] += acc;
  }
  __syncthreads();
  ln64(tid, P.lnf_g, P.lnf_b, xvec, xn, stats);

  // ---- head + mask + NLL ----
  float* logits = scores;
  if (tid < A){
    float acc = P.head_b[tid];
    for (int i = 0; i < D; ++i) acc += xn[i]*P.head_w[i*A + tid];
    if (P.mask[(size_t)b*A + tid] == 0) acc = -1e9f;
    logits[tid] = acc;
    P.out_logits[(size_t)b*A + tid] = acc;
  }
  __syncthreads();
  if (tid == 0){
    float m = -1e30f;
    for (int a = 0; a < A; ++a) m = fmaxf(m, logits[a]);
    float s = 0.f;
    for (int a = 0; a < A; ++a) s += __expf(logits[a] - m);
    int yy = P.y[b];
    P.losses[b] = -(logits[yy] - m - __logf(s));
  }
}

// Prepack sa_wk/sa_wv/ca_wk/ca_wv [k][n] fp32 -> [n][k] bf16 bits in ws.
__global__ void prepack_wkv(const float* sa_wk, const float* sa_wv,
                            const float* ca_wk, const float* ca_wv,
                            unsigned short* wbf){
  int idx = blockIdx.x*blockDim.x + threadIdx.x;
  if (idx >= 4*D*D) return;
  int mat = idx >> 12, r = idx & (D*D - 1);
  int n = r >> 6, k = r & 63;
  const float* src = (mat==0) ? sa_wk : (mat==1) ? sa_wv : (mat==2) ? ca_wk : ca_wv;
  wbf[idx] = f2bf(src[k*D + n]);
}

// Deterministic loss mean (fixed-order tree, no float atomics).
__global__ void loss_reduce(const float* losses, float* out_loss, int Bn){
  __shared__ float sm[256];
  float s = 0.f;
  for (int i = threadIdx.x; i < Bn; i += 256) s += losses[i];
  sm[threadIdx.x] = s; __syncthreads();
  for (int off = 128; off > 0; off >>= 1){
    if (threadIdx.x < off) sm[threadIdx.x] += sm[threadIdx.x + off];
    __syncthreads();
  }
  if (threadIdx.x == 0) *out_loss = sm[0] / (float)Bn;
}

extern "C" void kernel_launch(void* const* d_in, const int* in_sizes, int n_in,
                              void* d_out, int out_size, void* d_ws, size_t ws_size,
                              hipStream_t stream){
  enum {
    I_ENC=0, I_X1, I_X2, I_X3, I_MASK, I_Y,
    I_CONV1_W, I_CONV1_B, I_CONV2_W, I_CONV2_B,
    I_FC_X1_W, I_FC_X1_B, I_FC_M1_W, I_FC_M1_B, I_FC_M2_W, I_FC_M2_B,
    I_SA_WK, I_SA_BK, I_SA_WQ, I_SA_BQ, I_SA_WV, I_SA_BV, I_SA_PW, I_SA_PB,
    I_CA_WK, I_CA_BK, I_CA_WQ, I_CA_BQ, I_CA_WV, I_CA_BV, I_CA_PW, I_CA_PB,
    I_FF_W1, I_FF_B1, I_FF_W2, I_FF_B2, I_HEAD_W, I_HEAD_B,
    I_LN1_G, I_LN1_B, I_LN2_G, I_LN2_B, I_LN3_G, I_LN3_B, I_LNF_G, I_LNF_B
  };
  const int Bn = in_sizes[I_Y];
  auto F = [&](int i){ return (const float*)d_in[i]; };

  unsigned short* wbf = (unsigned short*)d_ws;
  float* losses = (float*)((char*)d_ws + (size_t)4*D*D*sizeof(unsigned short));

  prepack_wkv<<<(4*D*D + 255)/256, 256, 0, stream>>>(
      F(I_SA_WK), F(I_SA_WV), F(I_CA_WK), F(I_CA_WV), wbf);

  DecP P;
  P.enc = F(I_ENC); P.x1 = F(I_X1); P.x2 = F(I_X2); P.x3 = F(I_X3);
  P.mask = (const int*)d_in[I_MASK]; P.y = (const int*)d_in[I_Y];
  P.conv1_w = F(I_CONV1_W); P.conv1_b = F(I_CONV1_B);
  P.conv2_w = F(I_CONV2_W); P.conv2_b = F(I_CONV2_B);
  P.fc_x1_w = F(I_FC_X1_W); P.fc_x1_b = F(I_FC_X1_B);
  P.fc_m1_w = F(I_FC_M1_W); P.fc_m1_b = F(I_FC_M1_B);
  P.fc_m2_w = F(I_FC_M2_W); P.fc_m2_b = F(I_FC_M2_B);
  P.sa_wq = F(I_SA_WQ); P.sa_bq = F(I_SA_BQ); P.sa_bv = F(I_SA_BV);
  P.sa_pw = F(I_SA_PW); P.sa_pb = F(I_SA_PB);
  P.ca_wq = F(I_CA_WQ); P.ca_bq = F(I_CA_BQ); P.ca_bv = F(I_CA_BV);
  P.ca_pw = F(I_CA_PW); P.ca_pb = F(I_CA_PB);
  P.ff_w1 = F(I_FF_W1); P.ff_b1 = F(I_FF_B1);
  P.ff_w2 = F(I_FF_W2); P.ff_b2 = F(I_FF_B2);
  P.head_w = F(I_HEAD_W); P.head_b = F(I_HEAD_B);
  P.ln1_g = F(I_LN1_G); P.ln1_b = F(I_LN1_B);
  P.ln2_g = F(I_LN2_G); P.ln2_b = F(I_LN2_B);
  P.ln3_g = F(I_LN3_G); P.ln3_b = F(I_LN3_B);
  P.lnf_g = F(I_LNF_G); P.lnf_b = F(I_LNF_B);
  P.wbf = wbf;
  P.out_logits = (float*)d_out;
  P.losses = losses;

  decoder_fused<<<Bn, NTHREADS, 0, stream>>>(P, Bn);
  loss_reduce<<<1, 256, 0, stream>>>(losses, (float*)d_out + (size_t)Bn*A, Bn);
}
